// BiCell_31258771981059
// MI455X (gfx1250) — matrix operand order repaired and checked by reading, not compiled, not run on hardware
//
#include <hip/hip_runtime.h>

typedef __bf16 bf16;
typedef __attribute__((ext_vector_type(16))) __bf16 v16bf;
typedef __attribute__((ext_vector_type(8)))  __bf16 v8bf;
typedef __attribute__((ext_vector_type(8)))  float  v8f;

#define D_   512
#define CH_  2048
#define NB_  64
#define S_   512
#define GBLK 16   // workgroups per scan direction (128 waves = 512 tiles / 4)

__device__ inline v8f zero8() {
  v8f z = {0.f,0.f,0.f,0.f,0.f,0.f,0.f,0.f};
  return z;
}
// A fragment: per-lane chunks at +0..7 and +16..23 (16B each)
__device__ inline v16bf ldA(const bf16* p) {
  union { v16bf v; v8bf h[2]; } u;
  u.h[0] = *(const v8bf*)(p);
  u.h[1] = *(const v8bf*)(p + 16);
  return u.v;
}
// B fragment (weight stored N x K): per-lane chunks at +0..7 and +16..23 (16B each)
__device__ inline v16bf ldB(const bf16* p) {
  union { v16bf v; v8bf h[2]; } u;
  u.h[0] = *(const v8bf*)(p);
  u.h[1] = *(const v8bf*)(p + 16);
  return u.v;
}
// A fragment converted on the fly from f32 source
__device__ inline v16bf ldA_f32(const float* p) {
  union { v16bf v; bf16 e[16]; } u;
#pragma unroll
  for (int i = 0; i < 8; ++i) u.e[i] = (bf16)p[i];
#pragma unroll
  for (int i = 0; i < 8; ++i) u.e[8+i] = (bf16)p[16+i];
  return u.v;
}
__device__ inline v8f wmma_bf(v16bf a, v16bf b, v8f c) {
  return __builtin_amdgcn_wmma_f32_16x16x32_bf16(false, a, false, b, (short)0, c,
                                                 false, false);
}
__device__ inline float sigmoidf_(float x) { return 1.0f / (1.0f + __expf(-x)); }
__device__ inline float geluf_(float x) {
  return 0.5f * x * (1.0f + erff(x * 0.70710678118654752f));
}

// software grid barrier over one direction's GBLK blocks
__device__ inline void grid_barrier(unsigned* cnt, unsigned nblk, unsigned* expect) {
  __syncthreads();
  *expect += nblk;
  if (threadIdx.x == 0) {
    __threadfence();
    atomicAdd(cnt, 1u);
    volatile unsigned* vc = (volatile unsigned*)cnt;
    while (*vc < *expect) { __builtin_amdgcn_s_sleep(2); }
    __threadfence();
  }
  __syncthreads();
}

// ---- one-time weight convert+transpose: in (K,N) f32 -> out (N,K) bf16 ----
__global__ void __launch_bounds__(256) transconv_kernel(const float* __restrict__ in,
                                                        bf16* __restrict__ out,
                                                        int K, int N) {
  const int idx = blockIdx.x * 256 + threadIdx.x;
  if (idx >= K * N) return;
  const int n = idx / K, k = idx % K;
  out[idx] = (bf16)in[(size_t)k * N + n];
}

// ---- word projection + LayerNorm, fused; writes state in (S,N,D) layout ----
__global__ void __launch_bounds__(256) word_kernel(
    const float* __restrict__ x, const bf16* __restrict__ wT,
    const float* __restrict__ wb, const float* __restrict__ lng,
    const float* __restrict__ lnb, float* __restrict__ state_f,
    bf16* __restrict__ state_b) {
  __shared__ float sm[16 * 512];
  const int tid = threadIdx.x, wave = tid >> 5, lane = tid & 31;
  const int l = lane & 15, hi = lane >> 4;
  const int row0 = blockIdx.x * 16;
  v8f acc[4] = { zero8(), zero8(), zero8(), zero8() };
  const float* ar = x + (size_t)(row0 + l) * 512;
  for (int kb = 0; kb < 512; kb += 32) {
    v16bf af = ldA_f32(ar + kb + hi * 8);
#pragma unroll
    for (int t = 0; t < 4; ++t) {
      const int col = wave * 64 + t * 16 + l;
      acc[t] = wmma_bf(af, ldB(wT + (size_t)col * 512 + kb + hi * 8), acc[t]);
    }
  }
#pragma unroll
  for (int t = 0; t < 4; ++t) {
    const int col = wave * 64 + t * 16 + l;
    const float bias = wb[col];
#pragma unroll
    for (int v = 0; v < 8; ++v) sm[(hi * 8 + v) * 512 + col] = acc[t][v] + bias;
  }
  __syncthreads();
  for (int rr = wave * 2; rr < wave * 2 + 2; ++rr) {
    float s1 = 0.f, s2 = 0.f;
#pragma unroll
    for (int i = 0; i < 16; ++i) {
      float v = sm[rr * 512 + lane + i * 32];
      s1 += v; s2 += v * v;
    }
    for (int mg = 16; mg > 0; mg >>= 1) {
      s1 += __shfl_xor(s1, mg, 32);
      s2 += __shfl_xor(s2, mg, 32);
    }
    const float mean = s1 * (1.f / 512.f);
    const float var  = s2 * (1.f / 512.f) - mean * mean;
    const float inv  = rsqrtf(var + 1e-5f);
    const int rg = row0 + rr;
    const int n = rg >> 9, spos = rg & 511;          // x row = n*S + s
    const size_t dst = ((size_t)spos * NB_ + n) * D_;
#pragma unroll
    for (int i = 0; i < 16; ++i) {
      const int j = lane + i * 32;
      const float v = (sm[rr * 512 + j] - mean) * inv * lng[j] + lnb[j];
      state_f[dst + j] = v;
      state_b[dst + j] = (bf16)v;
    }
  }
}

// ---- init h carriers + zero barrier counters (every launch: deterministic) ----
__global__ void __launch_bounds__(256) init_kernel(
    float* hf, float* hb, bf16* hfb, bf16* hbb,
    const float* __restrict__ init_fh, const float* __restrict__ init_bh,
    unsigned* counters) {
  const int idx = blockIdx.x * 256 + threadIdx.x;
  if (blockIdx.x == 0 && threadIdx.x < 2) counters[threadIdx.x * 64] = 0u;
  if (idx < NB_ * D_) {
    const int dcol = idx & 511;
    const float vf = init_fh[dcol], vb = init_bh[dcol];
    hf[idx] = vf;  hb[idx] = vb;
    hfb[idx] = (bf16)vf;  hbb[idx] = (bf16)vb;
  }
}

// ---- persistent bidirectional scan: blocks [0,16)=fwd, [16,32)=bwd ----
// Each wave owns a 16x64 output strip (4 tiles): 1 A fragment feeds 4 WMMAs
// into 4 independent accumulator chains (fills the matrix pipe, 40B/lane/WMMA).
__global__ void __launch_bounds__(256) scan_kernel(
    const bf16* __restrict__ state_b, const float* __restrict__ state_f,
    const float* __restrict__ mask,
    const bf16* __restrict__ fw1T, const float* __restrict__ fb1,
    const bf16* __restrict__ fw2T, const float* __restrict__ fb2,
    const float* __restrict__ flng, const float* __restrict__ flnb,
    const bf16* __restrict__ bw1T, const float* __restrict__ bb1,
    const bf16* __restrict__ bw2T, const float* __restrict__ bb2,
    const float* __restrict__ blng, const float* __restrict__ blnb,
    float* hf, bf16* hfb, float* hb, bf16* hbb,
    bf16* it_f, bf16* it_b, float* ct_f, float* ct_b,
    bf16* fstore, bf16* bstore, unsigned* counters) {
  __shared__ float ps1[8];
  __shared__ float ps2[8];

  const int dir = blockIdx.x / GBLK;
  const int gid = blockIdx.x % GBLK;
  unsigned* cnt = counters + dir * 64;     // separate cachelines per direction
  unsigned expect = 0;

  const bf16*  w1T = dir ? bw1T : fw1T;
  const float* b1  = dir ? bb1  : fb1;
  const bf16*  w2T = dir ? bw2T : fw2T;
  const float* b2  = dir ? bb2  : fb2;
  const float* lng = dir ? blng : flng;
  const float* lnb = dir ? blnb : flnb;
  float* h   = dir ? hb   : hf;
  bf16*  hB  = dir ? hbb  : hfb;
  bf16*  itb = dir ? it_b : it_f;
  float* ct  = dir ? ct_b : ct_f;
  bf16*  st  = dir ? bstore : fstore;

  const int tid = threadIdx.x, wave = tid >> 5, lane = tid & 31;
  const int l = lane & 15, hi = lane >> 4;
  const int wid  = gid * 8 + wave;     // 0..127
  const int rowT = wid & 3;            // 4 row tiles (M=64)
  const int cg   = wid >> 2;           // 32 groups of 64 cols (N=2048)
  const int arow = rowT * 16 + l;

  // phase C mapping: 4 batch rows per block, 64 threads (2 waves) per row
  const int rquad = tid >> 6;          // 0..3
  const int rt    = tid & 63;          // thread within row

  for (int s = 0; s < S_; ++s) {
    const int pos = dir ? (S_ - 1 - s) : s;
    const bf16* inpB = state_b + (size_t)(pos * NB_) * D_;

    // GEMM1: it = gelu([h | state_pos] @ w1T + b1)   (K=1024)
    {
      v8f acc[4] = { zero8(), zero8(), zero8(), zero8() };
      for (int kb = 0; kb < 1024; kb += 32) {
        const bf16* ap = (kb < 512)
            ? (hB + (size_t)arow * 512 + kb + hi * 8)
            : (inpB + (size_t)arow * 512 + (kb - 512) + hi * 8);
        v16bf af = ldA(ap);
#pragma unroll
        for (int t = 0; t < 4; ++t) {
          const bf16* bp = w1T + (size_t)(cg * 64 + t * 16 + l) * 1024 + kb + hi * 8;
          acc[t] = wmma_bf(af, ldB(bp), acc[t]);
        }
      }
#pragma unroll
      for (int t = 0; t < 4; ++t) {
        const int col = cg * 64 + t * 16 + l;
        const float bias = b1[col];
#pragma unroll
        for (int v = 0; v < 8; ++v) {
          const int row = rowT * 16 + hi * 8 + v;
          itb[(size_t)row * CH_ + col] = (bf16)geluf_(acc[t][v] + bias);
        }
      }
    }
    grid_barrier(cnt, GBLK, &expect);

    // GEMM2: ct = it @ w2T + b2   (K=2048)
    {
      v8f acc[4] = { zero8(), zero8(), zero8(), zero8() };
      const bf16* ar = itb + (size_t)arow * CH_;
      for (int kb = 0; kb < 2048; kb += 32) {
        __builtin_prefetch(w2T + (size_t)(cg * 64 + l) * 2048 + kb + 512, 0, 1);
        v16bf af = ldA(ar + kb + hi * 8);
#pragma unroll
        for (int t = 0; t < 4; ++t) {
          const bf16* bp = w2T + (size_t)(cg * 64 + t * 16 + l) * 2048 + kb + hi * 8;
          acc[t] = wmma_bf(af, ldB(bp), acc[t]);
        }
      }
#pragma unroll
      for (int t = 0; t < 4; ++t) {
        const int col = cg * 64 + t * 16 + l;
        const float bias = b2[col];
#pragma unroll
        for (int v = 0; v < 8; ++v) {
          const int row = rowT * 16 + hi * 8 + v;
          ct[(size_t)row * 2048 + col] = acc[t][v] + bias;
        }
      }
    }
    grid_barrier(cnt, GBLK, &expect);

    // gates + mix + LayerNorm + mask blend; block handles rows gid*4..gid*4+3
    {
      const int n = gid * 4 + rquad;
      const float* cr  = ct + (size_t)n * 2048;
      const float* inF = state_f + (size_t)(pos * NB_ + n) * D_;
      float* hr = h + (size_t)n * D_;
      float mixv[8];
      float s1 = 0.f, s2 = 0.f;
#pragma unroll
      for (int i = 0; i < 8; ++i) {
        const int d = rt + i * 64;
        const float g0 = sigmoidf_(cr[d]);
        const float g1 = sigmoidf_(cr[512 + d]);
        const float g2 = sigmoidf_(cr[1024 + d]);
        mixv[i] = g0 * hr[d] + g1 * inF[d] + g2 * cr[1536 + d];
        s1 += mixv[i]; s2 += mixv[i] * mixv[i];
      }
      for (int mg = 16; mg > 0; mg >>= 1) {
        s1 += __shfl_xor(s1, mg, 32);
        s2 += __shfl_xor(s2, mg, 32);
      }
      if (lane == 0) { ps1[wave] = s1; ps2[wave] = s2; }
      __syncthreads();
      const float t1 = ps1[rquad * 2] + ps1[rquad * 2 + 1];
      const float t2 = ps2[rquad * 2] + ps2[rquad * 2 + 1];
      const float mean = t1 * (1.0f / 512.0f);
      const float var  = t2 * (1.0f / 512.0f) - mean * mean;
      const float inv  = rsqrtf(var + 1e-5f);
      const float m    = mask[(size_t)n * S_ + pos];
#pragma unroll
      for (int i = 0; i < 8; ++i) {
        const int d = rt + i * 64;
        const float hn = (mixv[i] - mean) * inv * lng[d] + lnb[d];
        const float nh = m * hn + (1.0f - m) * hr[d];
        hr[d] = nh;
        hB[(size_t)n * D_ + d] = (bf16)nh;
        st[((size_t)(pos * NB_ + n)) * D_ + d] = (bf16)nh;
      }
    }
    grid_barrier(cnt, GBLK, &expect);
  }
}

// ---- global_state = [h_f | h_b] @ c1_w + c1_b   (64x512, K=1024) ----
__global__ void __launch_bounds__(256) c1_kernel(
    const bf16* __restrict__ hfb, const bf16* __restrict__ hbb,
    const bf16* __restrict__ c1T, const float* __restrict__ c1b,
    float* __restrict__ outg) {
  const int tid = threadIdx.x, wave = tid >> 5, lane = tid & 31;
  const int l = lane & 15, hi = lane >> 4;
  const int wid = blockIdx.x * 8 + wave;      // 0..127
  const int rowT = wid & 3, colT = wid >> 2;  // colT 0..31
  const int arow = rowT * 16 + l, col = colT * 16 + l;
  v8f acc = zero8();
  const bf16* brow = c1T + (size_t)col * 1024;
  for (int kb = 0; kb < 1024; kb += 32) {
    const bf16* ap = (kb < 512) ? (hfb + (size_t)arow * 512 + kb + hi * 8)
                                : (hbb + (size_t)arow * 512 + (kb - 512) + hi * 8);
    acc = wmma_bf(ldA(ap), ldB(brow + kb + hi * 8), acc);
  }
  const float bias = c1b[col];
#pragma unroll
  for (int v = 0; v < 8; ++v) {
    const int row = rowT * 16 + hi * 8 + v;
    outg[(size_t)row * 512 + col] = acc[v] + bias;
  }
}

// ---- sequence = [fseq | bseq] @ c2_w + c2_b   (32768x512, K=1024) ----
__global__ void __launch_bounds__(256) c2_kernel(
    const bf16* __restrict__ fstore, const bf16* __restrict__ bstore,
    const bf16* __restrict__ c2T, const float* __restrict__ c2b,
    float* __restrict__ outseq) {
  const int tid = threadIdx.x, wave = tid >> 5, lane = tid & 31;
  const int l = lane & 15, hi = lane >> 4;
  const int wid  = blockIdx.x * 8 + wave;   // 0..16383
  const int rowT = wid >> 3;                // 0..2047
  const int cg   = wid & 7;                 // 64-col group (4 tiles/wave)
  const int r    = rowT * 16 + l;           // scan row = s*64 + n
  v8f acc[4] = { zero8(), zero8(), zero8(), zero8() };
  for (int kb = 0; kb < 1024; kb += 32) {
    const bf16* ap = (kb < 512) ? (fstore + (size_t)r * 512 + kb + hi * 8)
                                : (bstore + (size_t)r * 512 + (kb - 512) + hi * 8);
    v16bf af = ldA(ap);
#pragma unroll
    for (int t = 0; t < 4; ++t) {
      const bf16* bp = c2T + (size_t)(cg * 64 + t * 16 + l) * 1024 + kb + hi * 8;
      acc[t] = wmma_bf(af, ldB(bp), acc[t]);
    }
  }
#pragma unroll
  for (int t = 0; t < 4; ++t) {
    const int col = cg * 64 + t * 16 + l;
    const float bias = c2b[col];
#pragma unroll
    for (int v = 0; v < 8; ++v) {
      const int row = rowT * 16 + hi * 8 + v;   // = s*64 + n
      const int srow = row >> 6, n = row & 63;
      outseq[(size_t)n * (S_ * D_) + (size_t)srow * D_ + col] = acc[t][v] + bias;
    }
  }
}

extern "C" void kernel_launch(void* const* d_in, const int* in_sizes, int n_in,
                              void* d_out, int out_size, void* d_ws, size_t ws_size,
                              hipStream_t stream) {
  const float* x       = (const float*)d_in[0];
  const float* mask    = (const float*)d_in[1];
  const float* word_w  = (const float*)d_in[2];
  const float* word_b  = (const float*)d_in[3];
  const float* ln_g    = (const float*)d_in[4];
  const float* ln_b    = (const float*)d_in[5];
  const float* init_fh = (const float*)d_in[6];
  const float* init_bh = (const float*)d_in[7];
  const float* f_w1    = (const float*)d_in[8];
  const float* f_b1    = (const float*)d_in[9];
  const float* f_w2    = (const float*)d_in[10];
  const float* f_b2    = (const float*)d_in[11];
  const float* f_lng   = (const float*)d_in[12];
  const float* f_lnb   = (const float*)d_in[13];
  const float* b_w1    = (const float*)d_in[14];
  const float* b_b1    = (const float*)d_in[15];
  const float* b_w2    = (const float*)d_in[16];
  const float* b_b2    = (const float*)d_in[17];
  const float* b_lng   = (const float*)d_in[18];
  const float* b_lnb   = (const float*)d_in[19];
  const float* c1_w    = (const float*)d_in[20];
  const float* c1_b    = (const float*)d_in[21];
  const float* c2_w    = (const float*)d_in[22];
  const float* c2_b    = (const float*)d_in[23];

  char* ws = (char*)d_ws;
  size_t off = 0;
  auto alloc = [&](size_t bytes) -> void* {
    void* p = ws + off;
    off = (off + bytes + 255) & ~(size_t)255;
    return p;
  };
  unsigned* counters = (unsigned*)alloc(512);
  float* state_f = (float*)alloc((size_t)S_ * NB_ * D_ * 4);
  bf16*  state_b = (bf16*) alloc((size_t)S_ * NB_ * D_ * 2);
  bf16*  fstore  = (bf16*) alloc((size_t)S_ * NB_ * D_ * 2);
  bf16*  bstore  = (bf16*) alloc((size_t)S_ * NB_ * D_ * 2);
  bf16*  wordwT  = (bf16*) alloc((size_t)512 * 512 * 2);
  bf16*  fw1T    = (bf16*) alloc((size_t)2048 * 1024 * 2);
  bf16*  fw2T    = (bf16*) alloc((size_t)2048 * 2048 * 2);
  bf16*  bw1T    = (bf16*) alloc((size_t)2048 * 1024 * 2);
  bf16*  bw2T    = (bf16*) alloc((size_t)2048 * 2048 * 2);
  bf16*  c1T     = (bf16*) alloc((size_t)512 * 1024 * 2);
  bf16*  c2T     = (bf16*) alloc((size_t)512 * 1024 * 2);
  float* hf      = (float*)alloc((size_t)NB_ * D_ * 4);
  float* hb      = (float*)alloc((size_t)NB_ * D_ * 4);
  bf16*  hfb     = (bf16*) alloc((size_t)NB_ * D_ * 2);
  bf16*  hbb     = (bf16*) alloc((size_t)NB_ * D_ * 2);
  bf16*  it_f    = (bf16*) alloc((size_t)NB_ * CH_ * 2);
  bf16*  it_b    = (bf16*) alloc((size_t)NB_ * CH_ * 2);
  float* ct_f    = (float*)alloc((size_t)NB_ * 2048 * 4);
  float* ct_b    = (float*)alloc((size_t)NB_ * 2048 * 4);

  float* out_seq  = (float*)d_out;
  float* out_glob = out_seq + (size_t)NB_ * S_ * D_;

  auto tc = [&](const float* in, bf16* out, int K, int N) {
    int total = K * N;
    transconv_kernel<<<(total + 255) / 256, 256, 0, stream>>>(in, out, K, N);
  };
  tc(word_w, wordwT, 512, 512);
  tc(f_w1, fw1T, 1024, 2048);
  tc(f_w2, fw2T, 2048, 2048);
  tc(b_w1, bw1T, 1024, 2048);
  tc(b_w2, bw2T, 2048, 2048);
  tc(c1_w, c1T, 1024, 512);
  tc(c2_w, c2T, 1024, 512);

  word_kernel<<<(NB_ * S_) / 16, 256, 0, stream>>>(x, wordwT, word_b, ln_g, ln_b,
                                                   state_f, state_b);
  init_kernel<<<128, 256, 0, stream>>>(hf, hb, hfb, hbb, init_fh, init_bh, counters);
  scan_kernel<<<2 * GBLK, 256, 0, stream>>>(
      state_b, state_f, mask,
      fw1T, f_b1, fw2T, f_b2, f_lng, f_lnb,
      bw1T, b_b1, bw2T, b_b2, b_lng, b_lnb,
      hf, hfb, hb, hbb, it_f, it_b, ct_f, ct_b, fstore, bstore, counters);
  c1_kernel<<<16, 256, 0, stream>>>(hfb, hbb, c1T, c1_b, out_glob);
  c2_kernel<<<2048, 256, 0, stream>>>(fstore, bstore, c2T, c2_b, out_seq);

  (void)in_sizes; (void)n_in; (void)out_size; (void)ws_size;
}
